// GPT_86715389706669
// MI455X (gfx1250) — matrix operand order repaired and checked by reading, not compiled, not run on hardware
//
#include <hip/hip_runtime.h>
#include <cstdint>

#define DEVI __device__ __forceinline__

typedef __attribute__((ext_vector_type(16))) __bf16 v16bf;
typedef __attribute__((ext_vector_type(8)))  float  v8f;
typedef __attribute__((ext_vector_type(4))) unsigned u32x4;
typedef __attribute__((ext_vector_type(8))) int      i32x8;
typedef __attribute__((ext_vector_type(4))) int      i32x4;

static constexpr int B_ = 8, W_ = 512, CBLK_ = 24, C_ = 512, H_ = 8, L_ = 2, VOCAB_ = 256;
static constexpr int HD_ = C_ / H_;       // 64
static constexpr int NBW = B_ * W_;       // 4096 words
static constexpr int NCH = NBW * CBLK_;   // 98304 char rows
static constexpr int TC  = 3 * C_;        // 1536

DEVI unsigned short f2bf(float f) {
  unsigned u = __builtin_bit_cast(unsigned, f);
  u += 0x7FFFu + ((u >> 16) & 1u);        // round-to-nearest-even
  return (unsigned short)(u >> 16);
}
DEVI float bf2f(unsigned short h) {
  unsigned u = ((unsigned)h) << 16;
  return __builtin_bit_cast(float, u);
}
DEVI float wred_max(float v) {
#pragma unroll
  for (int o = 16; o > 0; o >>= 1) v = fmaxf(v, __shfl_xor(v, o, 32));
  return v;
}
DEVI float wred_sum(float v) {
#pragma unroll
  for (int o = 16; o > 0; o >>= 1) v += __shfl_xor(v, o, 32);
  return v;
}

union Frag { v16bf v; unsigned u[8]; };

// ---------------------------------------------------------------------------
// CDNA5 TDM helper: issue a 2D tensor DMA (global -> LDS) with optional LDS
// row padding. Descriptor bitfields per ISA 8.3/8.4 (D# group0/group1).
// data_size_code: 0=1B,1=2B,2=4B,3=8B. Dims/strides in data_size units.
// pad_interval code: interval = 2^(code+1) DWORDs; pad_amount code: (code+1) DWORDs.
// ---------------------------------------------------------------------------
DEVI void tdm_load_2d(unsigned lds_off, const void* gptr, unsigned data_size_code,
                      unsigned tensor_d0, unsigned tensor_d1, unsigned long long stride0,
                      unsigned tile_d0, unsigned tile_d1,
                      unsigned pad_interval, unsigned pad_amount, bool pad_en)
{
  unsigned long long ga = (unsigned long long)gptr;
  u32x4 g0; i32x8 g1; i32x4 g2 = (i32x4)0, g3 = (i32x4)0;
  g0[0] = 1u;                                             // count=1, user mode
  g0[1] = lds_off;                                        // lds_addr [63:32]
  g0[2] = (unsigned)(ga & 0xFFFFFFFFu);                   // global_addr [95:64]
  g0[3] = (unsigned)((ga >> 32) & 0x01FFFFFFu) | (2u << 30);  // addr[56:32] | type=2
  g1[0] = (int)((data_size_code << 16) | (pad_en ? (1u << 20) : 0u) |
                (pad_interval << 22) | (pad_amount << 25));
  g1[1] = (int)((tensor_d0 & 0xFFFFu) << 16);             // tensor_dim0[15:0] @ 63:48
  g1[2] = (int)(((tensor_d0 >> 16) & 0xFFFFu) | ((tensor_d1 & 0xFFFFu) << 16));
  g1[3] = (int)(((tensor_d1 >> 16) & 0xFFFFu) | ((tile_d0 & 0xFFFFu) << 16));
  g1[4] = (int)(tile_d1 & 0xFFFFu);                       // tile_dim1; tile_dim2=0
  g1[5] = (int)(unsigned)(stride0 & 0xFFFFFFFFu);         // dim0_stride[31:0]
  g1[6] = (int)(unsigned)((stride0 >> 32) & 0xFFFFu);     // dim0_stride[47:32]
  g1[7] = 0;
#if __has_builtin(__builtin_amdgcn_tensor_load_to_lds)
#if __clang_major__ >= 23
  i32x8 g4 = (i32x8)0;
  __builtin_amdgcn_tensor_load_to_lds(g0, g1, g2, g3, g4, 0);
#else
  __builtin_amdgcn_tensor_load_to_lds(g0, g1, g2, g3, 0);
#endif
#endif
}

DEVI void tdm_wait0() {
#if __has_builtin(__builtin_amdgcn_s_wait_tensorcnt)
  __builtin_amdgcn_s_wait_tensorcnt(0);
#else
  asm volatile("s_wait_tensorcnt 0x0" ::: "memory");
#endif
}

// ---------------------------------------------------------------------------
// WMMA GEMM: D[M,N] = act(A_bf16[M,K] @ Bw_f32[K,N] (+ R)) ; store f32/bf16
// Block tile 128x64, K-step 32; 8 waves in a 4(M)x2(N) grid, each wave owns a
// 32x32 output (2 A-frags x 2 B-frags -> 4 WMMAs per K-step).
// A tile is copied global->LDS with async-to-LDS (no VGPR round trip).
// All M % 128 == 0, N % 64 == 0, K % 32 == 0 in this model.
// ---------------------------------------------------------------------------
template<bool RESID, bool GELU, bool STOREF, bool STOREH>
__global__ __launch_bounds__(256)
void k_gemm(const unsigned short* __restrict__ A, const float* __restrict__ Bw,
            float* Df, unsigned short* Dh, const float* R, int M, int N, int K)
{
  constexpr int BM = 128, BN = 64, BK = 32, LDT = 40;   // LDS row stride (bf16 elems)
  __shared__ unsigned short sA[BM * LDT];               // [m][k]
  __shared__ unsigned short sB[BN * LDT];               // [n][k] (transposed)
  const int tid = threadIdx.x, wid = tid >> 5, lane = tid & 31;
  const int half = lane >> 4, r16 = lane & 15;
  const int wm = (wid & 3) * 32, wn = (wid >> 2) * 32;  // wave offsets in tile
  const int bm = blockIdx.x, bn = blockIdx.y;

  // cooperative-load index precompute
  const int arr = tid >> 1, ac0 = (tid & 1) << 4;       // A: 16 bf16 per thread
  const int bkr = tid >> 3, bn0 = (tid & 7) << 3;       // B: 8 f32 per thread
  const unsigned aldso = (unsigned)(size_t)(sA + arr * LDT + ac0);  // LDS byte offset

  v8f acc[2][2];
#pragma unroll
  for (int i = 0; i < 2; ++i)
#pragma unroll
    for (int j = 0; j < 2; ++j)
#pragma unroll
      for (int e = 0; e < 8; ++e) acc[i][j][e] = 0.f;

  for (int kk = 0; kk < K; kk += BK) {
    __syncthreads();
    { // A tile: 128x32 bf16, async copy 32B/thread straight into LDS
      const unsigned short* ap = A + (size_t)(bm * BM + arr) * K + kk + ac0;
      unsigned long long ga = (unsigned long long)ap;
      asm volatile("global_load_async_to_lds_b128 %0, %1, off"
                   :: "v"(aldso), "v"(ga) : "memory");
      asm volatile("global_load_async_to_lds_b128 %0, %1, off offset:16"
                   :: "v"(aldso), "v"(ga) : "memory");
      if (kk + BK < K) __builtin_prefetch(ap + BK, 0, 1);   // global_prefetch_b8
    }
    { // B tile: 32x64 f32 -> bf16 transposed [n][k], 8 f32 per thread
      const float* bp = Bw + (size_t)(kk + bkr) * N + bn * BN + bn0;
      float4 b0 = *(const float4*)bp;
      float4 b1 = *(const float4*)(bp + 4);
      sB[(bn0 + 0) * LDT + bkr] = f2bf(b0.x);
      sB[(bn0 + 1) * LDT + bkr] = f2bf(b0.y);
      sB[(bn0 + 2) * LDT + bkr] = f2bf(b0.z);
      sB[(bn0 + 3) * LDT + bkr] = f2bf(b0.w);
      sB[(bn0 + 4) * LDT + bkr] = f2bf(b1.x);
      sB[(bn0 + 5) * LDT + bkr] = f2bf(b1.y);
      sB[(bn0 + 6) * LDT + bkr] = f2bf(b1.z);
      sB[(bn0 + 7) * LDT + bkr] = f2bf(b1.w);
      if (kk + BK < K) __builtin_prefetch(bp + (size_t)BK * N, 0, 1);
    }
    asm volatile("s_wait_asynccnt 0x0" ::: "memory");       // A tile landed in LDS
    __syncthreads();
    Frag fa[2], fb[2];
#pragma unroll
    for (int j = 0; j < 8; ++j) {
      // 16-bit A 16x32 layout: vgpr j holds K = ((j>>2)*16) + half*8 + 2*(j&3)
      const int ka = ((j >> 2) << 4) + (half << 3) + ((j & 3) << 1);
      fa[0].u[j] = *(const unsigned*)&sA[(wm + r16) * LDT + ka];
      fa[1].u[j] = *(const unsigned*)&sA[(wm + 16 + r16) * LDT + ka];
      // B 32x16: same K order as A: vgpr j holds K = ((j>>2)*16) + half*8 + 2*(j&3)
      const int kb = ((j >> 2) << 4) + (half << 3) + ((j & 3) << 1);
      fb[0].u[j] = *(const unsigned*)&sB[(wn + r16) * LDT + kb];
      fb[1].u[j] = *(const unsigned*)&sB[(wn + 16 + r16) * LDT + kb];
    }
#pragma unroll
    for (int tm = 0; tm < 2; ++tm)
#pragma unroll
      for (int tn = 0; tn < 2; ++tn)
        acc[tm][tn] = __builtin_amdgcn_wmma_f32_16x16x32_bf16(
            false, fa[tm].v, false, fb[tn].v, (short)0, acc[tm][tn], false, false);
  }
  // C/D layout: vgpr v -> M = v + 8*half, N = lane&15
#pragma unroll
  for (int tm = 0; tm < 2; ++tm)
#pragma unroll
    for (int tn = 0; tn < 2; ++tn) {
      const int col = bn * BN + wn + tn * 16 + r16;
#pragma unroll
      for (int v = 0; v < 8; ++v) {
        const int row = bm * BM + wm + tm * 16 + v + (half << 3);
        const size_t o = (size_t)row * N + col;
        float x = acc[tm][tn][v];
        if (RESID) x += R[o];
        if (GELU)  x = 0.5f * x * (1.0f + erff(x * 0.70710678118654752f));
        if (STOREF) Df[o] = x;
        if (STOREH) Dh[o] = f2bf(x);
      }
    }
}

// ---------------------------------------------------------------------------
// LayerNorm over C=512, one wave per row, bf16 output (feeds WMMA GEMMs)
// ---------------------------------------------------------------------------
__global__ __launch_bounds__(256)
void k_ln(const float* __restrict__ X, const float* __restrict__ g,
          const float* __restrict__ b, unsigned short* __restrict__ Y, int rows)
{
  const int wid = threadIdx.x >> 5, lane = threadIdx.x & 31;
  const int row = blockIdx.x * 8 + wid;
  if (row >= rows) return;
  const float* x = X + (size_t)row * C_;
  float v[16]; float s = 0.f, sq = 0.f;
#pragma unroll
  for (int i = 0; i < 16; ++i) { float t = x[i * 32 + lane]; v[i] = t; s += t; sq += t * t; }
  s = wred_sum(s); sq = wred_sum(sq);
  const float mean = s * (1.f / C_);
  const float var  = sq * (1.f / C_) - mean * mean;
  const float rstd = rsqrtf(var + 1e-5f);
  unsigned short* y = Y + (size_t)row * C_;
#pragma unroll
  for (int i = 0; i < 16; ++i) {
    const int d = i * 32 + lane;
    y[d] = f2bf((v[i] - mean) * rstd * g[d] + b[d]);
  }
}

__global__ void k_lastix(const int* __restrict__ mask, int* __restrict__ li)
{
  int bw = blockIdx.x * blockDim.x + threadIdx.x;
  if (bw >= NBW) return;
  int s = 0;
#pragma unroll
  for (int j = 0; j < CBLK_; ++j) s += mask[bw * CBLK_ + j];
  li[bw] = s - 1;
}

__global__ void k_embed(const int* __restrict__ x, const float* __restrict__ cte,
                        const float* __restrict__ cpe, float* __restrict__ emb, int total)
{
  int idx = blockIdx.x * blockDim.x + threadIdx.x;
  if (idx >= total) return;
  int d = idx & (C_ - 1);
  int bwc = idx >> 9;            // C_ == 512
  int c = bwc % CBLK_;
  int tok = x[bwc];
  emb[idx] = cte[tok * C_ + d] + cpe[c * C_ + d];
}

__global__ void k_splice(const unsigned short* __restrict__ ctx,
                         const unsigned short* __restrict__ hln,
                         const int* __restrict__ li,
                         unsigned short* __restrict__ seq, int total)
{
  int idx = blockIdx.x * blockDim.x + threadIdx.x;
  if (idx >= total) return;
  int d = idx & (C_ - 1);
  int bwc = idx >> 9;
  int bw = bwc / CBLK_, c = bwc % CBLK_;
  seq[idx] = (c == li[bw]) ? hln[bw * C_ + d] : ctx[idx];
}

__global__ void k_tobf16(const float* __restrict__ X, unsigned short* __restrict__ Y, int n)
{
  int i = blockIdx.x * blockDim.x + threadIdx.x;
  if (i < n) Y[i] = f2bf(X[i]);
}

// ---------------------------------------------------------------------------
// Char attention: only query row last_ix matters (output is gathered there).
// One wave per (word, head): lane j = key j (<=23), softmax across lanes,
// then lanes own 2 dims each for the PV reduction. Optional +wpe (layer 0).
// ---------------------------------------------------------------------------
__global__ __launch_bounds__(256)
void k_char_attn(const float* __restrict__ qkv, const int* __restrict__ li,
                 const float* __restrict__ wpe, unsigned short* __restrict__ obuf,
                 int addwpe)
{
  __shared__ float ps[8][33];
  const int wid = threadIdx.x >> 5, lane = threadIdx.x & 31;
  const int gid = blockIdx.x * 8 + wid;       // grid is exact: NBW*H/8 blocks
  const int bw = gid >> 3, h = gid & 7;
  const int lw = li[bw];
  const float* base = qkv + (size_t)bw * CBLK_ * TC;
  const float* qrow = base + (size_t)lw * TC + h * HD_;
  float s = -3.4e38f;
  const bool valid = (lane < CBLK_) && (lane <= lw);
  if (valid) {
    const float* kr = base + (size_t)lane * TC + C_ + h * HD_;
    float a = 0.f;
#pragma unroll
    for (int d = 0; d < HD_; ++d) a += qrow[d] * kr[d];
    s = a * 0.125f;                            // 1/sqrt(64)
  }
  const float mx  = wred_max(s);
  const float p   = valid ? __expf(s - mx) : 0.f;
  const float sum = wred_sum(p);
  ps[wid][lane] = p;
  __syncthreads();
  const float inv = 1.f / sum;
  const int d0 = lane * 2;
  float o0 = 0.f, o1 = 0.f;
  const int jmax = (lw < CBLK_ - 1) ? lw : (CBLK_ - 1);
  for (int j = 0; j <= jmax; ++j) {
    const float pj = ps[wid][j];
    const float* vr = base + (size_t)j * TC + 2 * C_ + h * HD_ + d0;
    o0 += pj * vr[0]; o1 += pj * vr[1];
  }
  o0 *= inv; o1 *= inv;
  if (addwpe) {
    const int w = bw & (W_ - 1);
    o0 += wpe[w * C_ + h * HD_ + d0];
    o1 += wpe[w * C_ + h * HD_ + d0 + 1];
  }
  obuf[(size_t)bw * C_ + h * HD_ + d0]     = f2bf(o0);
  obuf[(size_t)bw * C_ + h * HD_ + d0 + 1] = f2bf(o1);
}

// ---------------------------------------------------------------------------
// Word attention: one block per (b,h); all K/V (512x64 f32, padded stride 68)
// staged in ~278KB LDS (WGP has 320KB) via TWO TDM tensor loads with LDS
// padding (interval 64 DW, pad 4 DW -> stride 68 floats). Online softmax,
// 1 query per wave.
// ---------------------------------------------------------------------------
__global__ __launch_bounds__(256)
void k_word_attn(const float* __restrict__ qkv, unsigned short* __restrict__ out)
{
  extern __shared__ float sm[];
  constexpr int KS = 68;                       // padded row stride (floats)
  float* Ksh = sm;                             // 512*68
  float* Vsh = Ksh + W_ * KS;                  // 512*68
  float* qsc = Vsh + W_ * KS;                  // 8*64 per-wave q broadcast
  float* psc = qsc + 8 * HD_;                  // 8*32 per-wave probs
  const int tid = threadIdx.x, wid = tid >> 5, lane = tid & 31;
  const int b = blockIdx.x >> 3, h = blockIdx.x & 7;
  if (wid == 0) {
    // TDM: tile 64(x) x 512(y) f32 from rows of qkv[b,:,:] into padded LDS
    const float* kg = qkv + (size_t)(b * W_) * TC + C_ + h * HD_;
    const float* vg = qkv + (size_t)(b * W_) * TC + 2 * C_ + h * HD_;
    tdm_load_2d((unsigned)(size_t)Ksh, kg, /*ds=*/2u, /*td0=*/TC, /*td1=*/W_,
                /*stride0=*/(unsigned long long)TC, /*tile0=*/HD_, /*tile1=*/W_,
                /*pad_int=*/5u, /*pad_amt=*/3u, /*pad_en=*/true);
    tdm_load_2d((unsigned)(size_t)Vsh, vg, 2u, TC, W_,
                (unsigned long long)TC, HD_, W_, 5u, 3u, true);
    tdm_wait0();
  }
  __syncthreads();
  float* qw = qsc + wid * HD_;
  float* pw = psc + wid * 32;
  for (int i = wid; i < W_; i += 8) {
    const size_t qb = ((size_t)(b * W_ + i)) * TC + h * HD_;
    qw[2 * lane]     = qkv[qb + 2 * lane];      // per-wave LDS, in-order DS ops
    qw[2 * lane + 1] = qkv[qb + 2 * lane + 1];
    float m = -3.4e38f, l = 0.f, o0 = 0.f, o1 = 0.f;
    for (int j0 = 0; j0 <= i; j0 += 32) {
      const int j = j0 + lane;
      const bool valid = (j <= i);
      float s = -3.4e38f;
      if (valid) {
        float a = 0.f;
        const float* kr = Ksh + j * KS;
#pragma unroll 8
        for (int d = 0; d < HD_; ++d) a += qw[d] * kr[d];
        s = a * 0.125f;
      }
      const float cm = wred_max(s);
      const float mn = fmaxf(m, cm);
      const float f  = __expf(m - mn);          // m=-3.4e38 first pass -> f=0
      const float p  = valid ? __expf(s - mn) : 0.f;
      const float cs = wred_sum(p);
      l = l * f + cs; o0 *= f; o1 *= f; m = mn;
      pw[lane] = p;
      const int jn = ((i - j0) < 31) ? (i - j0) : 31;
      for (int jj = 0; jj <= jn; ++jj) {
        const float pj = pw[jj];
        o0 += pj * Vsh[(j0 + jj) * KS + 2 * lane];
        o1 += pj * Vsh[(j0 + jj) * KS + 2 * lane + 1];
      }
    }
    const float invl = 1.f / l;
    const size_t ob = ((size_t)(b * W_ + i)) * C_ + h * HD_ + 2 * lane;
    out[ob]     = f2bf(o0 * invl);
    out[ob + 1] = f2bf(o1 * invl);
  }
}

// ---------------------------------------------------------------------------
extern "C" void kernel_launch(void* const* d_in, const int* in_sizes, int n_in,
                              void* d_out, int out_size, void* d_ws, size_t ws_size,
                              hipStream_t stream)
{
  (void)in_sizes; (void)n_in; (void)out_size; (void)ws_size;
  const int*   x       = (const int*)d_in[0];
  const int*   amask   = (const int*)d_in[1];
  const float* cte     = (const float*)d_in[2];
  const float* cpe     = (const float*)d_in[3];
  const float* wpe     = (const float*)d_in[4];
  const float* ln1g    = (const float*)d_in[5];
  const float* ln1b    = (const float*)d_in[6];
  const float* ln2g    = (const float*)d_in[7];
  const float* ln2b    = (const float*)d_in[8];
  const float* ln3g    = (const float*)d_in[9];
  const float* ln3b    = (const float*)d_in[10];
  const float* cattn_w    = (const float*)d_in[11];
  const float* cattn_proj = (const float*)d_in[12];
  const float* wattn_w    = (const float*)d_in[13];
  const float* wattn_proj = (const float*)d_in[14];
  const float* mlp_fc     = (const float*)d_in[15];
  const float* mlp_proj   = (const float*)d_in[16];
  const float* proj_w     = (const float*)d_in[17];
  const float* lm_head    = (const float*)d_in[18];
  float* out = (float*)d_out;

  char* ws = (char*)d_ws;
  size_t off = 0;
  auto alloc = [&](size_t bytes) -> char* {
    char* p = ws + off; off = (off + bytes + 255) & ~(size_t)255; return p;
  };
  int*            li    = (int*)           alloc((size_t)NBW * 4);
  float*          emb   = (float*)         alloc((size_t)NCH * C_ * 4);
  unsigned short* ctxh  = (unsigned short*)alloc((size_t)NCH * C_ * 2);
  unsigned short* seqh  = (unsigned short*)alloc((size_t)NCH * C_ * 2);
  float*          qkvc  = (float*)         alloc((size_t)NCH * TC * 4);
  unsigned short* obufh = (unsigned short*)alloc((size_t)NBW * C_ * 2);
  float*          state = (float*)         alloc((size_t)NBW * C_ * 4);
  unsigned short* lnth  = (unsigned short*)alloc((size_t)NBW * C_ * 2);
  unsigned short* hln1h = (unsigned short*)alloc((size_t)NBW * C_ * 2);
  float*          qkvw  = (float*)         alloc((size_t)NBW * TC * 4);
  unsigned short* woh   = (unsigned short*)alloc((size_t)NBW * C_ * 2);
  unsigned short* mlph  = (unsigned short*)alloc((size_t)NBW * 4 * C_ * 2);
  unsigned short* sth   = (unsigned short*)alloc((size_t)NBW * C_ * 2);
  unsigned short* yh    = (unsigned short*)alloc((size_t)NCH * C_ * 2);

  k_lastix<<<(NBW + 255) / 256, 256, 0, stream>>>(amask, li);
  { int total = NCH * C_;
    k_embed<<<(total + 255) / 256, 256, 0, stream>>>(x, cte, cpe, emb, total); }

  const size_t shw = (size_t)(2 * W_ * 68 + 8 * HD_ + 8 * 32) * 4;  // ~281.6 KB

  for (int l = 0; l < L_; ++l) {
    const unsigned short* seqA;
    if (l == 0) {
      k_ln<<<NCH / 8, 256, 0, stream>>>(emb, ln1g + l * C_, ln1b + l * C_, ctxh, NCH);
      seqA = ctxh;                               // cache context for layer 1
    } else {
      k_ln<<<NBW / 8, 256, 0, stream>>>(state, ln1g + l * C_, ln1b + l * C_, hln1h, NBW);
      int total = NCH * C_;
      k_splice<<<(total + 255) / 256, 256, 0, stream>>>(ctxh, hln1h, li, seqh, total);
      seqA = seqh;
    }
    // char qkv: [NCH,C] @ [C,3C]
    k_gemm<false,false,true,false><<<dim3(NCH / 128, TC / 64), 256, 0, stream>>>(
        seqA, cattn_w + (size_t)l * C_ * TC, qkvc, nullptr, nullptr, NCH, TC, C_);
    k_char_attn<<<NBW * H_ / 8, 256, 0, stream>>>(qkvc, li, wpe, obufh, l == 0 ? 1 : 0);
    // state = o @ cattn_proj  (no residual, per reference)
    k_gemm<false,false,true,false><<<dim3(NBW / 128, C_ / 64), 256, 0, stream>>>(
        obufh, cattn_proj + (size_t)l * C_ * C_, state, nullptr, nullptr, NBW, C_, C_);
    // word attention block
    k_ln<<<NBW / 8, 256, 0, stream>>>(state, ln2g + l * C_, ln2b + l * C_, lnth, NBW);
    k_gemm<false,false,true,false><<<dim3(NBW / 128, TC / 64), 256, 0, stream>>>(
        lnth, wattn_w + (size_t)l * C_ * TC, qkvw, nullptr, nullptr, NBW, TC, C_);
    k_word_attn<<<B_ * H_, 256, shw, stream>>>(qkvw, woh);
    k_gemm<true,false,true,false><<<dim3(NBW / 128, C_ / 64), 256, 0, stream>>>(
        woh, wattn_proj + (size_t)l * C_ * C_, state, nullptr, state, NBW, C_, C_);
    // MLP block
    k_ln<<<NBW / 8, 256, 0, stream>>>(state, ln3g + l * C_, ln3b + l * C_, lnth, NBW);
    k_gemm<false,true,false,true><<<dim3(NBW / 128, (4 * C_) / 64), 256, 0, stream>>>(
        lnth, mlp_fc + (size_t)l * C_ * 4 * C_, nullptr, mlph, nullptr, NBW, 4 * C_, C_);
    k_gemm<true,false,true,false><<<dim3(NBW / 128, C_ / 64), 256, 0, stream>>>(
        mlph, mlp_proj + (size_t)l * 4 * C_ * C_, state, nullptr, state, NBW, C_, 4 * C_);
  }

  { int n = NBW * C_;
    k_tobf16<<<(n + 255) / 256, 256, 0, stream>>>(state, sth, n); }
  // y = state @ proj_w  -> bf16 [NBW, 24*C] == [NCH, C]
  k_gemm<false,false,false,true><<<dim3(NBW / 128, (CBLK_ * C_) / 64), 256, 0, stream>>>(
      sth, proj_w, nullptr, yh, nullptr, NBW, CBLK_ * C_, C_);
  // logits = y @ lm_head -> f32 [NCH, VOCAB]
  k_gemm<false,false,true,false><<<dim3(NCH / 128, VOCAB_ / 64), 256, 0, stream>>>(
      yh, lm_head, out, nullptr, nullptr, NCH, VOCAB_, C_);
}
